// PointCloudEncoder_22110491639853
// MI455X (gfx1250) — matrix operand order repaired and checked by reading, not compiled, not run on hardware
//
#include <hip/hip_runtime.h>
#include <hip/hip_bf16.h>

// ---------------------------------------------------------------------------
// DGCNN point-cloud encoder for MI455X (gfx1250, wave32, WMMA).
// B=128 clouds, P=512 pts, K=10 knn, C=64 channels.
// EdgeConv trick: h_ij = [x_i, x_j-x_i] @ W = (A_i - Bm_i) + Bm_j where
//   A = X @ W_top, Bm = X @ W_bot  -> two P x 64 x 64 WMMA GEMMs per layer.
// kNN: s_ij = x_i.x_j - 0.5||x_j||^2 has the same ordering as top_k(-dist).
// New this round: HBM->LDS staging uses GLOBAL_LOAD_ASYNC_TO_LDS_B128
// (ASYNCcnt path) instead of global_load + ds_store.
// ---------------------------------------------------------------------------

#define NB   128   // clouds
#define NP   512   // points per cloud
#define NK   10    // knn k
#define CH   64    // padded feature channels
#define PB   8     // bbox corners
#define KB   4     // bbox knn k

typedef _Float16 half_t;
typedef __attribute__((ext_vector_type(16))) _Float16 v16h;
typedef __attribute__((ext_vector_type(8)))  _Float16 v8h;
typedef __attribute__((ext_vector_type(8)))  float    v8f;

// ---- CDNA5 async global->LDS copy (16B per lane), ASYNCcnt-tracked ---------
// VFLAT GV form: vdst = LDS byte address, vaddr = 64-bit global address, no SADDR.
// Low 32 bits of a generic shared pointer are the LDS byte offset (aperture
// maps addr[31:0] straight to LDS).
__device__ __forceinline__ void async_copy_16B(const void* gsrc, void* ldst) {
  unsigned int lds_off = (unsigned int)(unsigned long long)ldst;
  asm volatile("global_load_async_to_lds_b128 %0, %1, off"
               :: "v"(lds_off), "v"(gsrc) : "memory");
}
__device__ __forceinline__ void async_wait_all() {
  asm volatile("s_wait_asynccnt 0x0" ::: "memory");
}

// ---- WMMA fragment loaders (wave32; contiguous k order, not the ISA table order, same for A and B) ----------------------
// A (16x32 f16): lanes 0-15 -> M=lane, K=0..15;
//                lanes 16-31 -> M=lane-16, K=16..31.
__device__ __forceinline__ v16h load_a_frag(const half_t* base, int ld,
                                            int rowBase, int kBase, int lane) {
  int m    = lane & 15;
  int koff = (lane < 16) ? 0 : 16;
  return *(const v16h*)(base + (size_t)(rowBase + m) * ld + kBase + koff);
}

// B (32x16 f16), source stored N-major [n][k] so K is contiguous:
// lanes 0-15 -> N=lane, K=0..15; lanes 16-31 -> N=lane-16, K=16..31.
__device__ __forceinline__ v16h load_b_frag(const half_t* baseNmajor, int ld,
                                            int colBase, int kBase, int lane) {
  int n    = lane & 15;
  int koff = (lane < 16) ? 0 : 16;
  return *(const v16h*)(baseNmajor + (size_t)(colBase + n) * ld + kBase + koff);
}

__device__ __forceinline__ float selu_f(float x) {
  const float a = 1.6732632423543772f, s = 1.0507009873554805f;
  return s * (x > 0.f ? x : a * (__expf(x) - 1.f));
}

__device__ __forceinline__ float block64_sum(float v, volatile float* red, int tid) {
#pragma unroll
  for (int off = 16; off; off >>= 1) v += __shfl_xor(v, off, 32);
  if ((tid & 31) == 0) red[tid >> 5] = v;
  __syncthreads();
  float s = red[0] + red[1];
  __syncthreads();
  return s;
}

// ---- pad: f32 [B*P,3] coords -> f16 [B*P,64] zero-padded -------------------
__global__ void pad_coords_kernel(const float* __restrict__ x, half_t* __restrict__ X) {
  int i = blockIdx.x * blockDim.x + threadIdx.x;
  if (i >= NB * NP * CH) return;
  int c = i & (CH - 1), p = i >> 6;
  X[i] = (c < 3) ? (half_t)x[p * 3 + c] : (half_t)0.f;
}

// ---- knn: per (cloud, 64-row strip) WG; 4 waves x 16 rows ------------------
// LDS: X f16 (64KB) + sq f32 (2KB) + 4 strips of 16x512 f32 (128KB).
__global__ void __launch_bounds__(128) knn_kernel(const half_t* __restrict__ Xg,
                                                  int* __restrict__ idx) {
  extern __shared__ char smem_raw[];
  half_t* sX    = (half_t*)smem_raw;                     // 512*64 halves
  float*  sSq   = (float*)(smem_raw + 65536);            // 512
  float*  sStrip= (float*)(smem_raw + 65536 + 2048);     // 4 * 16*512

  int b = blockIdx.x >> 3, strip = blockIdx.x & 7;
  int tid = threadIdx.x;
  const half_t* Xc = Xg + (size_t)b * NP * CH;

  // async-stage the whole cloud's features into LDS (16B per lane per issue)
  for (int t = tid; t < NP * CH / 8; t += 128)
    async_copy_16B(Xc + t * 8, sX + t * 8);
  async_wait_all();
  __syncthreads();

  for (int r = tid; r < NP; r += 128) {
    float s = 0.f;
#pragma unroll
    for (int c = 0; c < CH; ++c) { float v = (float)sX[r * CH + c]; s += v * v; }
    sSq[r] = s;
  }
  __syncthreads();

  int wave = tid >> 5, lane = tid & 31;
  int rowBase = strip * 64 + wave * 16;
  float* strp = sStrip + wave * 16 * 512;

  v16h a0 = load_a_frag(sX, CH, rowBase, 0,  lane);
  v16h a1 = load_a_frag(sX, CH, rowBase, 32, lane);
  int n = lane & 15;
  int rowadd = (lane < 16) ? 0 : 8;

  for (int ct = 0; ct < 32; ++ct) {  // 32 column tiles of 16
    v8f acc = {};
    v16h b0 = load_b_frag(sX, CH, ct * 16, 0, lane);   // B = X^T (K-contig rows)
    acc = __builtin_amdgcn_wmma_f32_16x16x32_f16(false, a0, false, b0,
                                                 (short)0, acc, false, false);
    v16h b1 = load_b_frag(sX, CH, ct * 16, 32, lane);
    acc = __builtin_amdgcn_wmma_f32_16x16x32_f16(false, a1, false, b1,
                                                 (short)0, acc, false, false);
    int col = ct * 16 + n;
    float sj = -0.5f * sSq[col];
#pragma unroll
    for (int r = 0; r < 8; ++r)
      strp[(r + rowadd) * 512 + col] = acc[r] + sj;   // s = dot - 0.5*sq_j
  }

  // top-10 (max of s == min distance); lanes 0..15 each own one row.
  if (lane < 16) {
    float bv[NK]; int bi[NK];
#pragma unroll
    for (int t = 0; t < NK; ++t) { bv[t] = -3.4e38f; bi[t] = 0; }
    const float* row = strp + lane * 512;
    for (int j = 0; j < NP; ++j) {
      float v = row[j]; int vi = j;
#pragma unroll
      for (int t = 0; t < NK; ++t) {
        if (v > bv[t]) {
          float tv = bv[t]; bv[t] = v; v = tv;
          int   ti = bi[t]; bi[t] = vi; vi = ti;
        }
      }
    }
    int gr = b * NP + rowBase + lane;
#pragma unroll
    for (int t = 0; t < NK; ++t) idx[gr * NK + t] = bi[t];
  }
}

// ---- edgeconv: 1 WG (8 waves) per cloud ------------------------------------
// LDS: X f16 64KB | A f16 64KB | Bm f16 64KB | W^T f16 16KB | bias/g/beta.
__global__ void __launch_bounds__(256) edgeconv_kernel(
    const half_t* __restrict__ Xg, const int* __restrict__ idx,
    const float* __restrict__ W, const float* __restrict__ bias,
    const float* __restrict__ gamma, const float* __restrict__ beta,
    int Cin, half_t* __restrict__ Xout) {
  extern __shared__ char smem_raw[];
  half_t* sX   = (half_t*)smem_raw;
  half_t* sA   = (half_t*)(smem_raw + 65536);
  half_t* sBm  = (half_t*)(smem_raw + 131072);
  half_t* sWT  = (half_t*)(smem_raw + 196608);   // [mat][n][k] N-major, 2*64*64
  float*  sBias= (float*)(smem_raw + 212992);
  float*  sG   = sBias + 64;
  float*  sBe  = sBias + 128;

  int b = blockIdx.x, tid = threadIdx.x;
  const half_t* Xc = Xg + (size_t)b * NP * CH;

  // async-stage features; overlap with the W transpose/convert below.
  for (int t = tid; t < NP * CH / 8; t += 256)
    async_copy_16B(Xc + t * 8, sX + t * 8);

  // W (2*Cin x 64, row-major f32) -> transposed, zero-padded f16 in LDS
  for (int t = tid; t < 2 * 64 * 64; t += 256) {
    int mat = t >> 12, rem = t & 4095, nn = rem >> 6, kk = rem & 63;
    float v = (kk < Cin) ? W[(mat * Cin + kk) * CH + nn] : 0.f;
    sWT[t] = (half_t)v;
  }
  if (tid < 64) { sBias[tid] = bias[tid]; sG[tid] = gamma[tid]; sBe[tid] = beta[tid]; }
  async_wait_all();
  __syncthreads();

  int wave = tid >> 5, lane = tid & 31;
  int n = lane & 15, rowadd = (lane < 16) ? 0 : 8;

  // 256 output tiles (2 mats x 32 row-tiles x 4 col-tiles), 32 per wave.
  for (int tt = 0; tt < 32; ++tt) {
    int t = wave * 32 + tt;
    int mat = t >> 7, rem = t & 127, rt = rem >> 2, ctile = rem & 3;
    const half_t* wt = sWT + mat * 4096;
    half_t* dst = mat ? sBm : sA;
    v8f acc = {};
    v16h af0 = load_a_frag(sX, CH, rt * 16, 0, lane);
    v16h bf0 = load_b_frag(wt, CH, ctile * 16, 0, lane);
    acc = __builtin_amdgcn_wmma_f32_16x16x32_f16(false, af0, false, bf0,
                                                 (short)0, acc, false, false);
    v16h af1 = load_a_frag(sX, CH, rt * 16, 32, lane);
    v16h bf1 = load_b_frag(wt, CH, ctile * 16, 32, lane);
    acc = __builtin_amdgcn_wmma_f32_16x16x32_f16(false, af1, false, bf1,
                                                 (short)0, acc, false, false);
    int col = ctile * 16 + n;
#pragma unroll
    for (int r = 0; r < 8; ++r)
      dst[(rt * 16 + r + rowadd) * CH + col] = (half_t)acc[r];
  }
  __syncthreads();

  // Phase 2: LN + SELU + max over k neighbors. One wave per 64 points,
  // each lane owns channels {lane, lane+32}; mean/var via wave32 shuffles.
  int c0 = lane, c1 = lane + 32;
  const int* idxb = idx + (size_t)b * NP * NK;
  for (int pp = 0; pp < 64; ++pp) {
    int p = wave * 64 + pp;
    float base0 = (float)sA[p * CH + c0] - (float)sBm[p * CH + c0] + sBias[c0];
    float base1 = (float)sA[p * CH + c1] - (float)sBm[p * CH + c1] + sBias[c1];
    float mx0 = -3.4e38f, mx1 = -3.4e38f;
#pragma unroll
    for (int kk = 0; kk < NK; ++kk) {
      int j = idxb[p * NK + kk];
      float h0 = base0 + (float)sBm[j * CH + c0];
      float h1 = base1 + (float)sBm[j * CH + c1];
      float s = h0 + h1, q = h0 * h0 + h1 * h1;
#pragma unroll
      for (int off = 16; off; off >>= 1) {
        s += __shfl_xor(s, off, 32);
        q += __shfl_xor(q, off, 32);
      }
      float mean = s * (1.f / 64.f);
      float var  = q * (1.f / 64.f) - mean * mean;
      float rs   = rsqrtf(var + 1e-5f);
      mx0 = fmaxf(mx0, selu_f((h0 - mean) * rs * sG[c0] + sBe[c0]));
      mx1 = fmaxf(mx1, selu_f((h1 - mean) * rs * sG[c1] + sBe[c1]));
    }
    half_t* o = Xout + ((size_t)b * NP + p) * CH;
    o[c0] = (half_t)mx0;
    o[c1] = (half_t)mx1;
  }
}

// ---- per-cloud channel max pool -> g[b, layer*64 + c] ----------------------
__global__ void pool_kernel(const half_t* __restrict__ Xf, float* __restrict__ g,
                            int layer) {
  int b = blockIdx.x, c = threadIdx.x;  // 64 threads
  const half_t* p = Xf + (size_t)b * NP * CH + c;
  float m = -3.4e38f;
  for (int i = 0; i < NP; ++i) m = fmaxf(m, (float)p[i * CH]);
  g[b * 256 + layer * 64 + c] = m;
}

// ---- bbox encoder: tiny (8 pts), pure VALU, 64 threads / cloud -------------
__global__ void __launch_bounds__(64) bbox_kernel(
    const float* __restrict__ bbox,
    const float* W1, const float* b1, const float* g1, const float* be1,
    const float* W2, const float* b2, const float* g2, const float* be2,
    const float* Wp, const float* bp, const float* gp, const float* bep,
    float* __restrict__ gbb) {
  __shared__ float pts[PB][3];
  __shared__ int   nidx[PB][KB];
  __shared__ float f1[PB][CH];
  __shared__ float f2[PB][CH];
  __shared__ float red[2];
  __shared__ float dist[PB * PB];
  __shared__ float fb[128];

  int b = blockIdx.x, tid = threadIdx.x;
  if (tid < PB * 3) ((float*)pts)[tid] = bbox[b * PB * 3 + tid];
  __syncthreads();

  if (tid < PB) {  // knn over coords (self has -d = 0, always max)
    float bv[KB]; int bi[KB];
#pragma unroll
    for (int t = 0; t < KB; ++t) { bv[t] = -3.4e38f; bi[t] = 0; }
    for (int j = 0; j < PB; ++j) {
      float dx = pts[tid][0] - pts[j][0];
      float dy = pts[tid][1] - pts[j][1];
      float dz = pts[tid][2] - pts[j][2];
      float v = -(dx * dx + dy * dy + dz * dz); int vi = j;
#pragma unroll
      for (int t = 0; t < KB; ++t)
        if (v > bv[t]) { float tv = bv[t]; bv[t] = v; v = tv;
                         int   ti = bi[t]; bi[t] = vi; vi = ti; }
    }
#pragma unroll
    for (int t = 0; t < KB; ++t) nidx[tid][t] = bi[t];
  }
  __syncthreads();

  for (int i = 0; i < PB; ++i) {  // conv1 (Cin=3)
    float mx = -3.4e38f;
    for (int kk = 0; kk < KB; ++kk) {
      int j = nidx[i][kk];
      float h = b1[tid];
#pragma unroll
      for (int t = 0; t < 3; ++t) {
        h += pts[i][t] * W1[t * CH + tid];
        h += (pts[j][t] - pts[i][t]) * W1[(3 + t) * CH + tid];
      }
      float s = block64_sum(h, red, tid);
      float q = block64_sum(h * h, red, tid);
      float mean = s * (1.f / 64.f), var = q * (1.f / 64.f) - mean * mean;
      mx = fmaxf(mx, selu_f((h - mean) * rsqrtf(var + 1e-5f) * g1[tid] + be1[tid]));
    }
    f1[i][tid] = mx;
  }
  __syncthreads();

  {  // knn over f1 (64-dim), one (i,j) pair per thread
    int i = tid >> 3, j = tid & 7;
    float d = 0.f;
    for (int c = 0; c < CH; ++c) { float t = f1[i][c] - f1[j][c]; d += t * t; }
    dist[tid] = -d;
  }
  __syncthreads();
  if (tid < PB) {
    float bv[KB]; int bi[KB];
#pragma unroll
    for (int t = 0; t < KB; ++t) { bv[t] = -3.4e38f; bi[t] = 0; }
    for (int j = 0; j < PB; ++j) {
      float v = dist[tid * PB + j]; int vi = j;
#pragma unroll
      for (int t = 0; t < KB; ++t)
        if (v > bv[t]) { float tv = bv[t]; bv[t] = v; v = tv;
                         int   ti = bi[t]; bi[t] = vi; vi = ti; }
    }
#pragma unroll
    for (int t = 0; t < KB; ++t) nidx[tid][t] = bi[t];
  }
  __syncthreads();

  for (int i = 0; i < PB; ++i) {  // conv2 (Cin=64)
    float mx = -3.4e38f;
    for (int kk = 0; kk < KB; ++kk) {
      int j = nidx[i][kk];
      float h = b2[tid];
      for (int c = 0; c < CH; ++c) {
        h += f1[i][c] * W2[c * CH + tid];
        h += (f1[j][c] - f1[i][c]) * W2[(64 + c) * CH + tid];
      }
      float s = block64_sum(h, red, tid);
      float q = block64_sum(h * h, red, tid);
      float mean = s * (1.f / 64.f), var = q * (1.f / 64.f) - mean * mean;
      mx = fmaxf(mx, selu_f((h - mean) * rsqrtf(var + 1e-5f) * g2[tid] + be2[tid]));
    }
    f2[i][tid] = mx;
  }
  __syncthreads();

  {  // concat max over the 8 corners
    float m1 = -3.4e38f, m2 = -3.4e38f;
    for (int i = 0; i < PB; ++i) { m1 = fmaxf(m1, f1[i][tid]); m2 = fmaxf(m2, f2[i][tid]); }
    fb[tid] = m1; fb[64 + tid] = m2;
  }
  __syncthreads();

  // bb_proj: 128 -> 64, LN + SELU
  float h = bp[tid];
  for (int t = 0; t < 128; ++t) h += fb[t] * Wp[t * CH + tid];
  float s = block64_sum(h, red, tid);
  float q = block64_sum(h * h, red, tid);
  float mean = s * (1.f / 64.f), var = q * (1.f / 64.f) - mean * mean;
  gbb[b * 64 + tid] = selu_f((h - mean) * rsqrtf(var + 1e-5f) * gp[tid] + bep[tid]);
}

// ---- head: [g(256) | gbb(64)] @ Wf(320x128) + bf, L2 normalize -------------
__global__ void __launch_bounds__(128) head_kernel(
    const float* __restrict__ g, const float* __restrict__ gbb,
    const float* __restrict__ Wf, const float* __restrict__ bf,
    float* __restrict__ out) {
  __shared__ float feat[320];
  __shared__ float red4[4];
  int b = blockIdx.x, tid = threadIdx.x;
  for (int t = tid; t < 256; t += 128) feat[t] = g[b * 256 + t];
  if (tid < 64) feat[256 + tid] = gbb[b * 64 + tid];
  __syncthreads();
  float acc = bf[tid];
  for (int t = 0; t < 320; ++t) acc += feat[t] * Wf[t * 128 + tid];
  float q = acc * acc;
#pragma unroll
  for (int off = 16; off; off >>= 1) q += __shfl_xor(q, off, 32);
  if ((tid & 31) == 0) red4[tid >> 5] = q;
  __syncthreads();
  float ss = red4[0] + red4[1] + red4[2] + red4[3];
  out[b * 128 + tid] = acc / (sqrtf(ss) + 1e-9f);
}

// ---------------------------------------------------------------------------
extern "C" void kernel_launch(void* const* d_in, const int* in_sizes, int n_in,
                              void* d_out, int out_size, void* d_ws, size_t ws_size,
                              hipStream_t stream) {
  (void)in_sizes; (void)n_in; (void)out_size; (void)ws_size;
  // inputs in setup_inputs() dict order; params leaves in insertion order.
  const float* x    = (const float*)d_in[0];
  // d_in[1] = n_pts (unused; all clouds are full)
  const float* bbox = (const float*)d_in[2];
  const float *gcW[4], *gcB[4], *gcG[4], *gcBe[4];
  int i = 3;
  for (int l = 0; l < 4; ++l) {
    gcW[l] = (const float*)d_in[i];     gcB[l]  = (const float*)d_in[i + 1];
    gcG[l] = (const float*)d_in[i + 2]; gcBe[l] = (const float*)d_in[i + 3];
    i += 4;
  }
  const float *bb1W = (const float*)d_in[i],     *bb1B = (const float*)d_in[i + 1],
              *bb1G = (const float*)d_in[i + 2], *bb1E = (const float*)d_in[i + 3];
  i += 4;
  const float *bb2W = (const float*)d_in[i],     *bb2B = (const float*)d_in[i + 1],
              *bb2G = (const float*)d_in[i + 2], *bb2E = (const float*)d_in[i + 3];
  i += 4;
  const float *bbpW = (const float*)d_in[i],     *bbpB = (const float*)d_in[i + 1],
              *bbpG = (const float*)d_in[i + 2], *bbpE = (const float*)d_in[i + 3];
  i += 4;
  const float* Wf = (const float*)d_in[i];
  const float* bf = (const float*)d_in[i + 1];

  // workspace carve (~20 MB)
  char* ws = (char*)d_ws;
  size_t off = 0;
  half_t* Xa = (half_t*)(ws + off); off += (size_t)NB * NP * CH * sizeof(half_t);
  half_t* Xb = (half_t*)(ws + off); off += (size_t)NB * NP * CH * sizeof(half_t);
  int*    idx= (int*)   (ws + off); off += (size_t)NB * NP * NK * sizeof(int);
  float*  g  = (float*) (ws + off); off += (size_t)NB * 256 * sizeof(float);
  float*  gbb= (float*) (ws + off); off += (size_t)NB * 64 * sizeof(float);

  pad_coords_kernel<<<(NB * NP * CH + 255) / 256, 256, 0, stream>>>(x, Xa);

  const size_t knnShm = 65536 + 2048 + 4 * 16 * 512 * sizeof(float);   // ~194 KB
  const size_t ecShm  = 212992 + 3 * 64 * sizeof(float);               // ~209 KB

  half_t* cur = Xa; half_t* nxt = Xb;
  const int cins[4] = {3, 64, 64, 64};
  for (int l = 0; l < 4; ++l) {
    knn_kernel<<<NB * 8, 128, knnShm, stream>>>(cur, idx);
    edgeconv_kernel<<<NB, 256, ecShm, stream>>>(cur, idx, gcW[l], gcB[l], gcG[l],
                                                gcBe[l], cins[l], nxt);
    pool_kernel<<<NB, 64, 0, stream>>>(nxt, g, l);
    half_t* t = cur; cur = nxt; nxt = t;
  }

  bbox_kernel<<<NB, 64, 0, stream>>>(bbox,
                                     bb1W, bb1B, bb1G, bb1E,
                                     bb2W, bb2B, bb2G, bb2E,
                                     bbpW, bbpB, bbpG, bbpE, gbb);
  head_kernel<<<NB, 128, 0, stream>>>(g, gbb, Wf, bf, (float*)d_out);
}
